// GNNEncoder_7172595384376
// MI455X (gfx1250) — compile-verified, matrix operand order checked by reading, not run on hardware
//
#include <hip/hip_runtime.h>

typedef __attribute__((ext_vector_type(16))) _Float16 v16h;
typedef __attribute__((ext_vector_type(8)))  _Float16 v8h;
typedef __attribute__((ext_vector_type(8)))  float    v8f;

#define N_NODES 100000
#define N_EDGES 1280000
#define D_IN    64
#define D_HID   128
#define D_OUT   64

// ---------------------------------------------------------------------------
// Zero a float region (grid-stride).
// ---------------------------------------------------------------------------
__global__ void zero_kernel(float* __restrict__ p, long long n) {
    long long i = (long long)blockIdx.x * blockDim.x + threadIdx.x;
    long long stride = (long long)gridDim.x * blockDim.x;
    for (; i < n; i += stride) p[i] = 0.0f;
}

// ---------------------------------------------------------------------------
// f32 -> f16 elementwise convert (grid-stride).
// ---------------------------------------------------------------------------
__global__ void cvt_f16_kernel(const float* __restrict__ src,
                               _Float16* __restrict__ dst, long long n) {
    long long i = (long long)blockIdx.x * blockDim.x + threadIdx.x;
    long long stride = (long long)gridDim.x * blockDim.x;
    for (; i < n; i += stride) dst[i] = (_Float16)src[i];
}

// ---------------------------------------------------------------------------
// mean = agg * (1/max(cnt,1)), emitted directly as f16 (grid-stride).
// D is a power of two (64 or 128), but is passed as a runtime argument.
// ---------------------------------------------------------------------------
__global__ void mean_cvt_kernel(const float* __restrict__ agg,
                                const float* __restrict__ cnt,
                                _Float16* __restrict__ out,
                                long long total, int D) {
    long long i = (long long)blockIdx.x * blockDim.x + threadIdx.x;
    long long stride = (long long)gridDim.x * blockDim.x;
    for (; i < total; i += stride) {
        int node = (int)(i / D);
        float c = cnt[node];
        float invc = (c > 0.0f) ? (1.0f / c) : 0.0f;
        out[i] = (_Float16)(agg[i] * invc);
    }
}

// ---------------------------------------------------------------------------
// Layer-1 scatter: 16 threads per edge, each gathers one b128 (4 f32) of
// x[src] and atomically accumulates 4 f32 into agg[dst]. Thread q==0 also
// bumps the in-degree count.
// ---------------------------------------------------------------------------
__global__ void scatter_add_f32_kernel(const float* __restrict__ x,
                                       const long long* __restrict__ src,
                                       const long long* __restrict__ dst,
                                       float* __restrict__ agg,
                                       float* __restrict__ cnt) {
    long long tid = (long long)blockIdx.x * blockDim.x + threadIdx.x;
    long long total = (long long)N_EDGES * 16;
    if (tid >= total) return;
    int e = (int)(tid >> 4);
    int q = (int)(tid & 15);          // 4-float segment within the row
    long long s = src[e];
    long long d = dst[e];
    const float4 v = *(const float4*)(x + s * D_IN + 4 * q);
    float* a = agg + d * D_IN + 4 * q;
    atomicAdd(a + 0, v.x);
    atomicAdd(a + 1, v.y);
    atomicAdd(a + 2, v.z);
    atomicAdd(a + 3, v.w);
    if (q == 0) atomicAdd(&cnt[d], 1.0f);
}

// ---------------------------------------------------------------------------
// Layer-2 scatter: 16 threads per edge, each gathers one b128 (8 f16) of
// h[src] and atomically accumulates 8 f32 into agg[dst].
// ---------------------------------------------------------------------------
__global__ void scatter_add_f16_kernel(const _Float16* __restrict__ h,
                                       const long long* __restrict__ src,
                                       const long long* __restrict__ dst,
                                       float* __restrict__ agg) {
    long long tid = (long long)blockIdx.x * blockDim.x + threadIdx.x;
    long long total = (long long)N_EDGES * 16;
    if (tid >= total) return;
    int e = (int)(tid >> 4);
    int q = (int)(tid & 15);          // 8-half segment within the row
    long long s = src[e];
    long long d = dst[e];
    const v8h v = *(const v8h*)(h + s * D_HID + 8 * q);
    float* a = agg + d * D_HID + 8 * q;
#pragma unroll
    for (int j = 0; j < 8; ++j) atomicAdd(a + j, (float)v[j]);
}

// ---------------------------------------------------------------------------
// Fused SAGE layer via WMMA, all-f16 operands loaded directly with b128:
//   out[n, o] = act( Amean[n,:] @ Wl[o,:] + bl[o] + Aroot[n,:] @ Wr[o,:] )
// One wave computes a 16(node) x 16(out) tile; K-chunks of 32; two WMMAs per
// chunk share one f32 accumulator.
//
// A 16x32 f16 lane layout (ISA 7.12.2): lane l -> m = l&15, half = l>>4;
// elements [0..7]  = K = k0 + 8*half      .. +7 (one aligned 16B load)
// elements [8..15] = K = k0 + 16 + 8*half .. +7 (one aligned 16B load)
// B 32x16 mirrors this with the lane indexing the output column.
// C/D: vgpr r, lane l -> row = r + 8*(l>>4), col = l&15.
// ---------------------------------------------------------------------------
__global__ void sage_gemm_wmma_kernel(const _Float16* __restrict__ Amean, // [N,K]
                                      const _Float16* __restrict__ Aroot, // [N,K]
                                      const _Float16* __restrict__ WlH,   // [Dout,K]
                                      const _Float16* __restrict__ WrH,   // [Dout,K]
                                      const float* __restrict__ bl,       // [Dout]
                                      float* __restrict__ outF,           // [N,Dout] or null
                                      _Float16* __restrict__ outH,        // [N,Dout] or null
                                      int K, int Dout, int doRelu) {
    const int lane = threadIdx.x & 31;
    const int wave = threadIdx.x >> 5;
    const int tilesO = Dout >> 4;
    const int nTiles = (N_NODES >> 4) * tilesO;
    const int wid = blockIdx.x * (blockDim.x >> 5) + wave;
    if (wid >= nTiles) return;

    const int tileN = wid / tilesO;
    const int tileO = wid % tilesO;

    const int m    = lane & 15;
    const int half = lane >> 4;
    const int node = (tileN << 4) + m;   // this lane's node row (A side)
    const int ocol = (tileO << 4) + m;   // this lane's output column (B side)

    const _Float16* amRow = Amean + (long long)node * K;
    const _Float16* axRow = Aroot + (long long)node * K;
    const _Float16* wlRow = WlH  + (long long)ocol * K;
    const _Float16* wrRow = WrH  + (long long)ocol * K;

    const float bias = bl[(tileO << 4) + m];

    v8f acc = {};

    for (int k0 = 0; k0 < K; k0 += 32) {
        const int kLo = k0 + 8 * half;
        const int kHi = k0 + 16 + 8 * half;

        v8h amLo = *(const v8h*)(amRow + kLo);
        v8h amHi = *(const v8h*)(amRow + kHi);
        v8h axLo = *(const v8h*)(axRow + kLo);
        v8h axHi = *(const v8h*)(axRow + kHi);
        v8h wlLo = *(const v8h*)(wlRow + kLo);
        v8h wlHi = *(const v8h*)(wlRow + kHi);
        v8h wrLo = *(const v8h*)(wrRow + kLo);
        v8h wrHi = *(const v8h*)(wrRow + kHi);

        v16h aM = __builtin_shufflevector(amLo, amHi, 0, 1, 2, 3, 4, 5, 6, 7,
                                          8, 9, 10, 11, 12, 13, 14, 15);
        v16h aX = __builtin_shufflevector(axLo, axHi, 0, 1, 2, 3, 4, 5, 6, 7,
                                          8, 9, 10, 11, 12, 13, 14, 15);
        v16h bL = __builtin_shufflevector(wlLo, wlHi, 0, 1, 2, 3, 4, 5, 6, 7,
                                          8, 9, 10, 11, 12, 13, 14, 15);
        v16h bR = __builtin_shufflevector(wrLo, wrHi, 0, 1, 2, 3, 4, 5, 6, 7,
                                          8, 9, 10, 11, 12, 13, 14, 15);

        // mean-neighbor branch: C += A_mean x W_l^T
        acc = __builtin_amdgcn_wmma_f32_16x16x32_f16(false, aM, false, bL,
                                                     (short)0, acc, false, false);
        // root branch: C += A_root x W_r^T
        acc = __builtin_amdgcn_wmma_f32_16x16x32_f16(false, aX, false, bR,
                                                     (short)0, acc, false, false);
    }

    // Epilogue: bias (+ ReLU), write 16x16 tile (f32 and/or f16 destinations).
#pragma unroll
    for (int r = 0; r < 8; ++r) {
        const int row = (tileN << 4) + r + 8 * half;
        const int col = (tileO << 4) + m;
        float v = acc[r] + bias;
        if (doRelu) v = fmaxf(v, 0.0f);
        if (outF) outF[(long long)row * Dout + col] = v;
        if (outH) outH[(long long)row * Dout + col] = (_Float16)v;
    }
}

// ---------------------------------------------------------------------------
// Launcher
// ---------------------------------------------------------------------------
extern "C" void kernel_launch(void* const* d_in, const int* in_sizes, int n_in,
                              void* d_out, int out_size, void* d_ws, size_t ws_size,
                              hipStream_t stream) {
    (void)in_sizes; (void)n_in; (void)out_size; (void)ws_size;

    const float*     x    = (const float*)d_in[0];
    const long long* ei   = (const long long*)d_in[1];   // int64 edge_index [2, E]
    const float*     Wl1  = (const float*)d_in[2];       // [128, 64]
    const float*     bl1  = (const float*)d_in[3];       // [128]
    const float*     Wr1  = (const float*)d_in[4];       // [128, 64]
    const float*     Wl2  = (const float*)d_in[5];       // [64, 128]
    const float*     bl2  = (const float*)d_in[6];       // [64]
    const float*     Wr2  = (const float*)d_in[7];       // [64, 128]
    float*           out  = (float*)d_out;               // [N, 64] f32

    const long long* src = ei;
    const long long* dst = ei + N_EDGES;

    // ---- Workspace layout -------------------------------------------------
    // f32 region: cnt[N] + agg[N*128] (reused by both layers)
    // f16 region: xh[N*64] + mh[N*128] (mean, reused) + hh[N*128] + weights
    char* wsb = (char*)d_ws;
    float* cnt  = (float*)wsb;                                     // [N]
    float* agg  = cnt + N_NODES;                                   // [N,128] (layer1 uses [N,64])
    _Float16* xh  = (_Float16*)(agg + (long long)N_NODES * D_HID); // [N,64]
    _Float16* mh  = xh + (long long)N_NODES * D_IN;                // [N,128] mean (reused)
    _Float16* hh  = mh + (long long)N_NODES * D_HID;               // [N,128] hidden f16
    _Float16* wl1h = hh + (long long)N_NODES * D_HID;              // [128,64]
    _Float16* wr1h = wl1h + D_HID * D_IN;                          // [128,64]
    _Float16* wl2h = wr1h + D_HID * D_IN;                          // [64,128]
    _Float16* wr2h = wl2h + D_OUT * D_HID;                         // [64,128]

    // ---- 0) One-time f16 conversions (weights + x) ------------------------
    cvt_f16_kernel<<<4096, 256, 0, stream>>>(x, xh, (long long)N_NODES * D_IN);
    cvt_f16_kernel<<<32, 256, 0, stream>>>(Wl1, wl1h, (long long)D_HID * D_IN);
    cvt_f16_kernel<<<32, 256, 0, stream>>>(Wr1, wr1h, (long long)D_HID * D_IN);
    cvt_f16_kernel<<<32, 256, 0, stream>>>(Wl2, wl2h, (long long)D_OUT * D_HID);
    cvt_f16_kernel<<<32, 256, 0, stream>>>(Wr2, wr2h, (long long)D_OUT * D_HID);

    // ---- 1) Zero cnt + agg[N*64] for layer 1 ------------------------------
    zero_kernel<<<2048, 256, 0, stream>>>(cnt, (long long)N_NODES * (1 + D_IN));

    // ---- 2) Layer-1 scatter: agg += x[src], cnt += 1 ----------------------
    {
        const long long total = (long long)N_EDGES * 16;
        const int blocks = (int)((total + 255) / 256);
        scatter_add_f32_kernel<<<blocks, 256, 0, stream>>>(x, src, dst, agg, cnt);
    }

    // ---- 3) mean1 = agg/deg -> f16 ----------------------------------------
    mean_cvt_kernel<<<4096, 256, 0, stream>>>(agg, cnt, mh,
        (long long)N_NODES * D_IN, D_IN);

    // ---- 4) Layer-1 fused WMMA GEMM + bias + ReLU -> hh (f16) -------------
    {
        const int nTiles = (N_NODES / 16) * (D_HID / 16);   // 50000
        const int blocks = (nTiles + 7) / 8;                // 8 waves/block
        sage_gemm_wmma_kernel<<<blocks, 256, 0, stream>>>(mh, xh, wl1h, wr1h,
                                                          bl1, nullptr, hh,
                                                          D_IN, D_HID, 1);
    }

    // ---- 5) Re-zero agg[N*128] for layer 2 --------------------------------
    zero_kernel<<<4096, 256, 0, stream>>>(agg, (long long)N_NODES * D_HID);

    // ---- 6) Layer-2 scatter: agg += hh[src] (f16 b128 gather, f32 atomics) -
    {
        const long long total = (long long)N_EDGES * 16;
        const int blocks = (int)((total + 255) / 256);
        scatter_add_f16_kernel<<<blocks, 256, 0, stream>>>(hh, src, dst, agg);
    }

    // ---- 7) mean2 = agg/deg -> f16 ----------------------------------------
    mean_cvt_kernel<<<4096, 256, 0, stream>>>(agg, cnt, mh,
        (long long)N_NODES * D_HID, D_HID);

    // ---- 8) Layer-2 fused WMMA GEMM + bias -> out (f32) -------------------
    {
        const int nTiles = (N_NODES / 16) * (D_OUT / 16);   // 25000
        const int blocks = (nTiles + 7) / 8;
        sage_gemm_wmma_kernel<<<blocks, 256, 0, stream>>>(mh, hh, wl2h, wr2h,
                                                          bl2, out, nullptr,
                                                          D_HID, D_OUT, 0);
    }
}
